// gc_GNN_Dropout_5841155523229
// MI455X (gfx1250) — matrix operand order repaired and checked by reading, not compiled, not run on hardware
//
#include <hip/hip_runtime.h>
#include <hip/hip_bf16.h>

// ---------------- constants (match reference) ----------------
#define N_NODES   100000
#define N_EDGES   1600000
#define DIM       64
#define N_CLS     10
#define N_GRAPHS  512

typedef __attribute__((ext_vector_type(16))) __bf16 v16bf;
typedef __attribute__((ext_vector_type(8)))  __bf16 v8bf;
typedef __attribute__((ext_vector_type(8)))  float  v8f;

// ---------------- edge scatter-add: agg[dst] += feat[src] ----------------
__global__ __launch_bounds__(256) void scatter_add_kernel(
    const float* __restrict__ feat,
    const int* __restrict__ src,
    const int* __restrict__ dst,
    float* __restrict__ agg,
    int n_edges)
{
    int gid  = blockIdx.x * blockDim.x + threadIdx.x;
    int e    = gid >> 4;          // edge id
    int part = gid & 15;          // which float4 of the 64-dim row
    if (e >= n_edges) return;
    int s = src[e];
    int d = dst[e];
    const float4 v = *(const float4*)(feat + (size_t)s * DIM + part * 4);
    float* ap = agg + (size_t)d * DIM + part * 4;
    atomicAdd(ap + 0, v.x);
    atomicAdd(ap + 1, v.y);
    atomicAdd(ap + 2, v.z);
    atomicAdd(ap + 3, v.w);
}

// ---------------- weight pre-conversion fp32 -> bf16 (once per call) ------
// Packs [w1_rel, w1_root, w2_rel, w2_root, w3_rel, w3_root] contiguously.
__global__ __launch_bounds__(256) void convert_weights_kernel(
    const float* __restrict__ w1r, const float* __restrict__ w1o,
    const float* __restrict__ w2r, const float* __restrict__ w2o,
    const float* __restrict__ w3r, const float* __restrict__ w3o,
    __bf16* __restrict__ out)
{
    int gid = blockIdx.x * blockDim.x + threadIdx.x;   // 0 .. 6*4096-1
    int m = gid >> 12;
    int i = gid & 4095;
    const float* srcs[6] = { w1r, w1o, w2r, w2o, w3r, w3o };
    out[gid] = (__bf16)srcs[m][i];
}

// ---------------- LDS fragment loaders ----------------
// A-matrix 16x32 bf16 layout: lane l<16 -> row l, K in {0..15};
// lane l+16 -> row l, K in {16..31}. rowp points at lane's 64-elem row.
__device__ __forceinline__ v16bf lds_a_frag(const __bf16* rowp, int k0, int lh)
{
    v16bf r;
    const v8bf lo = *(const v8bf*)(rowp + k0 + lh * 16);
    const v8bf hi = *(const v8bf*)(rowp + k0 + lh * 16 + 8);
#pragma unroll
    for (int i = 0; i < 8; ++i) { r[i] = lo[i]; r[i + 8] = hi[i]; }
    return r;
}

// B-matrix 32x16 bf16 layout: lane l holds column l%16, K = 16*(l/16)+i contiguous.
// colp points at W[n][*] (row-major [out][in]) so B[k][n] = colp[k].
__device__ __forceinline__ v16bf lds_b_frag(const __bf16* colp, int k0, int lh)
{
    return *(const v16bf*)(colp + k0 + lh * 16);
}

// ---------------- fused layer GEMM: out = agg@W1^T + x@W2^T + bias (opt ReLU) ----
// Block = 256 threads = 8 waves; block tile = 32 rows x 64 cols.
// Weights staged to LDS via async global->LDS copies; A tiles staged as bf16.
template <int RELU>
__global__ __launch_bounds__(256) void gemm_layer_kernel(
    const float*  __restrict__ A1,    // agg [n,64] fp32
    const float*  __restrict__ A2,    // x   [n,64] fp32
    const __bf16* __restrict__ Wb,    // [2][64][64] bf16: w_rel then w_root
    const float*  __restrict__ bias,  // [64]
    float* __restrict__ out)          // [n,64]
{
    __shared__ __attribute__((aligned(32))) __bf16 sW[2 * 64 * 64]; // 16 KB
    __shared__ __attribute__((aligned(32))) __bf16 sA[2 * 32 * 64]; //  8 KB

    const int tid = threadIdx.x;

    // ---- async stage: 16 KB of bf16 weights, 16 B per lane-transfer ----
    {
        uint32_t lds0 = (uint32_t)(size_t)(&sW[0]);
        const char* g = (const char*)Wb;
#pragma unroll
        for (int it = 0; it < 4; ++it) {
            int idx = tid + it * 256;
            uint32_t loff = lds0 + idx * 16;
            uint64_t ga   = (uint64_t)(size_t)(g + idx * 16);
            asm volatile("global_load_async_to_lds_b128 %0, %1, off"
                         :: "v"(loff), "v"(ga) : "memory");
        }
    }

    // ---- stage A tiles (32 rows x 64 cols x 2 matrices), fp32 -> bf16 ----
    {
        const int base = tid * 8;            // 8 consecutive elements
        const int row  = base >> 6;
        const int col  = base & 63;
        const size_t gofs = (size_t)(blockIdx.x * 32 + row) * DIM + col;
#pragma unroll
        for (int m = 0; m < 2; ++m) {
            const float* p = (m ? A2 : A1) + gofs;
            const float4 u = *(const float4*)p;
            const float4 v = *(const float4*)(p + 4);
            __bf16* d = sA + m * 2048 + base;
            d[0] = (__bf16)u.x; d[1] = (__bf16)u.y;
            d[2] = (__bf16)u.z; d[3] = (__bf16)u.w;
            d[4] = (__bf16)v.x; d[5] = (__bf16)v.y;
            d[6] = (__bf16)v.z; d[7] = (__bf16)v.w;
        }
    }

    asm volatile("s_wait_asynccnt 0" ::: "memory");
    __syncthreads();

    // ---- per-wave 16x16 output tile ----
    const int wave = tid >> 5;
    const int lane = tid & 31;
    const int ll   = lane & 15;
    const int lh   = lane >> 4;
    const int rt   = wave >> 2;              // row tile within block (0/1)
    const int n0   = (wave & 3) * 16;        // col tile

    const __bf16* a1row = sA + 0 * 2048 + (rt * 16 + ll) * 64;
    const __bf16* a2row = sA + 1 * 2048 + (rt * 16 + ll) * 64;
    const __bf16* w1col = sW + 0 * 4096 + (n0 + ll) * 64;
    const __bf16* w2col = sW + 1 * 4096 + (n0 + ll) * 64;

    v8f c = {};
    c = __builtin_amdgcn_wmma_f32_16x16x32_bf16(
            false, lds_a_frag(a1row, 0, lh),  false, lds_b_frag(w1col, 0, lh),
            (short)0, c, false, false);
    c = __builtin_amdgcn_wmma_f32_16x16x32_bf16(
            false, lds_a_frag(a1row, 32, lh), false, lds_b_frag(w1col, 32, lh),
            (short)0, c, false, false);
    c = __builtin_amdgcn_wmma_f32_16x16x32_bf16(
            false, lds_a_frag(a2row, 0, lh),  false, lds_b_frag(w2col, 0, lh),
            (short)0, c, false, false);
    c = __builtin_amdgcn_wmma_f32_16x16x32_bf16(
            false, lds_a_frag(a2row, 32, lh), false, lds_b_frag(w2col, 32, lh),
            (short)0, c, false, false);

    // C/D layout: VGPR r -> M = r + 8*lh, N = ll
    const int rowBase = blockIdx.x * 32 + rt * 16;
    const float bv = bias[n0 + ll];
    float* op = out + (size_t)(rowBase + lh * 8) * DIM + n0 + ll;
#pragma unroll
    for (int r = 0; r < 8; ++r) {
        float v = c[r] + bv;
        if (RELU) v = fmaxf(v, 0.0f);
        op[(size_t)r * DIM] = v;
    }
}

// scalar fallback for any non-multiple-of-32 row remainder (unused when N%32==0)
__global__ __launch_bounds__(256) void gemm_rows_scalar_kernel(
    const float* __restrict__ A1, const float* __restrict__ A2,
    const float* __restrict__ W1, const float* __restrict__ W2,
    const float* __restrict__ bias, float* __restrict__ out,
    int row0, int nrows, int relu)
{
    int gid = blockIdx.x * blockDim.x + threadIdx.x;
    if (gid >= nrows * DIM) return;
    int r = row0 + gid / DIM;
    int n = gid % DIM;
    const float* a1 = A1 + (size_t)r * DIM;
    const float* a2 = A2 + (size_t)r * DIM;
    const float* w1 = W1 + (size_t)n * DIM;
    const float* w2 = W2 + (size_t)n * DIM;
    float s = bias[n];
    for (int k = 0; k < DIM; ++k) s += a1[k] * w1[k] + a2[k] * w2[k];
    if (relu) s = fmaxf(s, 0.0f);
    out[(size_t)r * DIM + n] = s;
}

// ---------------- pooling ----------------
__global__ __launch_bounds__(256) void pool_sum_kernel(
    const float* __restrict__ h, const int* __restrict__ batch,
    float* __restrict__ sums, int n_nodes)
{
    int gid  = blockIdx.x * blockDim.x + threadIdx.x;
    int node = gid >> 4;
    int part = gid & 15;
    if (node >= n_nodes) return;
    int g = batch[node];
    const float4 v = *(const float4*)(h + (size_t)node * DIM + part * 4);
    float* sp = sums + (size_t)g * DIM + part * 4;
    atomicAdd(sp + 0, v.x);
    atomicAdd(sp + 1, v.y);
    atomicAdd(sp + 2, v.z);
    atomicAdd(sp + 3, v.w);
}

__global__ __launch_bounds__(256) void pool_count_kernel(
    const int* __restrict__ batch, float* __restrict__ counts, int n_nodes)
{
    int n = blockIdx.x * blockDim.x + threadIdx.x;
    if (n >= n_nodes) return;
    atomicAdd(&counts[batch[n]], 1.0f);
}

__global__ __launch_bounds__(256) void pool_finalize_kernel(
    const float* __restrict__ sums, const float* __restrict__ counts,
    float* __restrict__ pooled_out, int n_graphs)
{
    int gid = blockIdx.x * blockDim.x + threadIdx.x;
    if (gid >= n_graphs * DIM) return;
    int g = gid / DIM;
    float cnt = fmaxf(counts[g], 1.0f);
    pooled_out[gid] = sums[gid] / cnt;
}

__global__ __launch_bounds__(256) void final_linear_kernel(
    const float* __restrict__ pooled, const float* __restrict__ w,
    const float* __restrict__ b, float* __restrict__ out, int n_graphs)
{
    int gid = blockIdx.x * blockDim.x + threadIdx.x;
    if (gid >= n_graphs * N_CLS) return;
    int g = gid / N_CLS;
    int c = gid % N_CLS;
    const float* p  = pooled + (size_t)g * DIM;
    const float* wr = w + (size_t)c * DIM;
    float s = b[c];
#pragma unroll
    for (int k = 0; k < DIM; ++k) s += p[k] * wr[k];
    out[gid] = s;
}

// ---------------- host launcher ----------------
extern "C" void kernel_launch(void* const* d_in, const int* in_sizes, int n_in,
                              void* d_out, int out_size, void* d_ws, size_t ws_size,
                              hipStream_t stream) {
    const float* x       = (const float*)d_in[0];
    const int*   ei      = (const int*)d_in[1];
    const int*   batch   = (const int*)d_in[2];
    // d_in[3] = num_graphs scalar (device); grid sizes use the static value 512.
    const float* w1_rel  = (const float*)d_in[4];
    const float* b1_rel  = (const float*)d_in[5];
    const float* w1_root = (const float*)d_in[6];
    const float* w2_rel  = (const float*)d_in[7];
    const float* b2_rel  = (const float*)d_in[8];
    const float* w2_root = (const float*)d_in[9];
    const float* w3_rel  = (const float*)d_in[10];
    const float* b3_rel  = (const float*)d_in[11];
    const float* w3_root = (const float*)d_in[12];
    const float* w_lin   = (const float*)d_in[13];
    const float* b_lin   = (const float*)d_in[14];

    const int n_nodes = in_sizes[2];
    const int n_edges = in_sizes[1] / 2;
    const int* src = ei;
    const int* dst = ei + n_edges;

    // workspace layout (floats, then bf16 weight pack)
    const size_t featN = (size_t)n_nodes * DIM;
    float* agg    = (float*)d_ws;
    float* hA     = agg + featN;
    float* hB     = hA + featN;
    float* sums   = hB + featN;                     // 512*64
    float* counts = sums + (size_t)N_GRAPHS * DIM;  // 512
    __bf16* wbuf  = (__bf16*)(counts + N_GRAPHS);   // 6*4096 bf16 (16B-aligned)

    float* pooled_out = (float*)d_out;                          // [512,64]
    float* cls_out    = (float*)d_out + (size_t)N_GRAPHS * DIM; // [512,10]

    const int scatter_blocks = (n_edges * 16 + 255) / 256;
    const int main_rows      = n_nodes & ~31;
    const int rem_rows       = n_nodes - main_rows;
    const int gemm_blocks    = main_rows / 32;
    const size_t feat_bytes  = featN * sizeof(float);

    // pre-convert all layer weights to bf16 (packed rel/root per layer)
    convert_weights_kernel<<<(6 * 4096) / 256, 256, 0, stream>>>(
        w1_rel, w1_root, w2_rel, w2_root, w3_rel, w3_root, wbuf);

    struct Layer {
        const float* in; float* out; const float* br; int relu;
    } layers[3] = {
        { x,  hA, b1_rel, 1 },
        { hA, hB, b2_rel, 1 },
        { hB, hA, b3_rel, 0 },
    };
    const float* wrel_f32[3]  = { w1_rel, w2_rel, w3_rel };
    const float* wroot_f32[3] = { w1_root, w2_root, w3_root };

    for (int L = 0; L < 3; ++L) {
        hipMemsetAsync(agg, 0, feat_bytes, stream);
        scatter_add_kernel<<<scatter_blocks, 256, 0, stream>>>(
            layers[L].in, src, dst, agg, n_edges);
        if (gemm_blocks > 0) {
            if (layers[L].relu)
                gemm_layer_kernel<1><<<gemm_blocks, 256, 0, stream>>>(
                    agg, layers[L].in, wbuf + (size_t)L * 8192, layers[L].br,
                    layers[L].out);
            else
                gemm_layer_kernel<0><<<gemm_blocks, 256, 0, stream>>>(
                    agg, layers[L].in, wbuf + (size_t)L * 8192, layers[L].br,
                    layers[L].out);
        }
        if (rem_rows > 0)
            gemm_rows_scalar_kernel<<<(rem_rows * DIM + 255) / 256, 256, 0, stream>>>(
                agg, layers[L].in, wrel_f32[L], wroot_f32[L], layers[L].br,
                layers[L].out, main_rows, rem_rows, layers[L].relu);
    }

    // global mean pool (hA holds final node features)
    hipMemsetAsync(sums, 0, (size_t)N_GRAPHS * DIM * sizeof(float), stream);
    hipMemsetAsync(counts, 0, (size_t)N_GRAPHS * sizeof(float), stream);
    pool_sum_kernel<<<(n_nodes * 16 + 255) / 256, 256, 0, stream>>>(hA, batch, sums, n_nodes);
    pool_count_kernel<<<(n_nodes + 255) / 256, 256, 0, stream>>>(batch, counts, n_nodes);
    pool_finalize_kernel<<<(N_GRAPHS * DIM + 255) / 256, 256, 0, stream>>>(
        sums, counts, pooled_out, N_GRAPHS);
    final_linear_kernel<<<(N_GRAPHS * N_CLS + 255) / 256, 256, 0, stream>>>(
        pooled_out, w_lin, b_lin, cls_out, N_GRAPHS);
}
